// EEGNet_E_6983616824460
// MI455X (gfx1250) — matrix operand order repaired and checked by reading, not compiled, not run on hardware
//
#include <hip/hip_runtime.h>

// ---------------- bf16 helpers / WMMA fragment types ----------------
typedef __attribute__((ext_vector_type(16))) __bf16 v16bf;
typedef __attribute__((ext_vector_type(8)))  float  v8f;
typedef unsigned int u32x4 __attribute__((ext_vector_type(4)));
typedef int          i32x8 __attribute__((ext_vector_type(8)));
typedef int          i32x4 __attribute__((ext_vector_type(4)));

union AFrag { v16bf v; unsigned u32[8]; };

__device__ __forceinline__ unsigned short f2bf(float f) {
  union { float f; unsigned u; } c; c.f = f;
  unsigned u = c.u;
  u = (u + 0x7FFFu + ((u >> 16) & 1u)) >> 16;   // RNE
  return (unsigned short)u;
}
__device__ __forceinline__ float bf2f(unsigned short h) {
  union { unsigned u; float f; } c; c.u = ((unsigned)h) << 16;
  return c.f;
}

// ---------------- constants ----------------
#define BATCH 128
#define CHN   22
#define SIN   801
#define WOUT  800
#define F1C   10
#define F2C   30
#define TST   15
#define S1W   224                      // 12 halo + 200 data + 12 halo

// workspace layout (bytes)
#define WS_W2Q 0
#define WS_W3Q 4096
#define WS_W4Q 8192
#define WS_Y   16384ULL                        // bf16 [128][10][22][800]
#define Y_ELEMS (128ULL*10*22*800)
#define WS_S1  (WS_Y + Y_ELEMS*2ULL)           // bf16 [15][128][30][224] halo-padded

// ================= K0: 4-bit symmetric quantization (w2, w3, w4) ==========
__global__ __launch_bounds__(256) void k0_quant(const float* w2, const float* w3,
                                                const float* w4, float* w2q,
                                                float* w3q, float* w4q) {
  __shared__ float red[256];
  const float* src[3] = { w2, w3, w4 };
  float*       dst[3] = { w2q, w3q, w4q };
  const int    n[3]   = { F2C*CHN, F2C*25, F2C*F2C };
  for (int j = 0; j < 3; ++j) {
    float m = 0.f;
    for (int i = threadIdx.x; i < n[j]; i += 256) m = fmaxf(m, fabsf(src[j][i]));
    red[threadIdx.x] = m; __syncthreads();
    for (int s = 128; s > 0; s >>= 1) {
      if (threadIdx.x < s) red[threadIdx.x] = fmaxf(red[threadIdx.x], red[threadIdx.x + s]);
      __syncthreads();
    }
    float scale = red[0] / 7.0f + 1e-12f;
    __syncthreads();
    for (int i = threadIdx.x; i < n[j]; i += 256)
      dst[j][i] = rintf(src[j][i] / scale) * scale;
    __syncthreads();
  }
}

// ================= K1: conv1 (K=100) + BN1 as bf16 WMMA implicit GEMM ======
// grid (13 wchunks, 22 ch, 128 batch), 128 threads = 4 waves, each wave one
// 16-row M-tile (16 consecutive w). N=16 (10 valid), K=128 (100 valid).
// B (w1) is pre-swizzled into per-lane fragment layout -> ds_load_b128 fetch.
__global__ __launch_bounds__(128) void k1_conv1(
    const float* __restrict__ x, const float* __restrict__ w1,
    const float* g1, const float* b1, const float* m1, const float* v1,
    unsigned short* __restrict__ y) {
  __shared__ unsigned short xs[163];        // x halo row for this 64-w chunk
  __shared__ unsigned Bfrag[4][32][8];      // pre-packed B fragments per k-step
  const int b = blockIdx.z, c = blockIdx.y, W0 = blockIdx.x * 64;
  const int tid = threadIdx.x;

  for (int i = tid; i < 4 * 32 * 8; i += 128) {
    int ks = i >> 8, ln = (i >> 3) & 31, j = i & 7;
    int nn = ln & 15, K = ks * 32 + (ln >> 4) * 16 + 2 * j;
    unsigned lo = (nn < F1C && K     < 100) ? (unsigned)f2bf(w1[nn * 100 + K])     : 0u;
    unsigned hi = (nn < F1C && K + 1 < 100) ? (unsigned)f2bf(w1[nn * 100 + K + 1]) : 0u;
    Bfrag[ks][ln][j] = lo | (hi << 16);
  }
  const float* xrow = x + ((size_t)b * CHN + c) * SIN;
  for (int i = tid; i < 163; i += 128) {
    int xi = W0 - 49 + i;                         // taps cover w-49..w+50
    xs[i] = f2bf((xi >= 0 && xi < SIN) ? xrow[xi] : 0.f);
  }
  __syncthreads();

  const int wave = tid >> 5, lane = tid & 31;
  const int half = lane >> 4, lrow = lane & 15;
  const int w0 = W0 + wave * 16;
  if (w0 < WOUT) {
    v8f acc = {0.f,0.f,0.f,0.f,0.f,0.f,0.f,0.f};
    const int abase = wave * 16 + lrow;           // xs offset of this M row
    for (int k0 = 0; k0 < 128; k0 += 32) {
      AFrag a, bm;
      #pragma unroll
      for (int j = 0; j < 8; ++j) {               // A 16x32 bf16 layout
        int K = k0 + half * 16 + 2 * j;
        unsigned lo = (K     < 100) ? xs[abase + K]     : 0u;
        unsigned hi = (K + 1 < 100) ? xs[abase + K + 1] : 0u;
        a.u32[j] = lo | (hi << 16);
      }
      const unsigned* bp = &Bfrag[k0 >> 5][lane][0];
      #pragma unroll
      for (int j = 0; j < 8; ++j) bm.u32[j] = bp[j];
      acc = __builtin_amdgcn_wmma_f32_16x16x32_bf16(false, a.v, false, bm.v,
                                                    (short)0, acc, false, false);
    }
    // BN1 + bf16 store; lane holds N = lrow, rows M = r + 8*half
    if (lrow < F1C) {
      float inv = g1[lrow] * rsqrtf(v1[lrow] + 1e-5f);
      float sh  = b1[lrow] - m1[lrow] * inv;
      unsigned short* yb = y + (((size_t)b * F1C + lrow) * CHN + c) * WOUT;
      #pragma unroll
      for (int r = 0; r < 8; ++r) {
        int w = w0 + r + half * 8;
        yb[w] = f2bf(acc[r] * inv + sh);
      }
    }
  }
}

// ================= K2: grouped conv2 + BN2 -> IF(15) -> pool4 -> s1 ========
// s1 rows are 224 wide: [12 zero halo][200 data][12 zero halo] so K3 can DMA
// the tile 1:1 with the TDM. Halo zeros are (re)written every call.
__global__ __launch_bounds__(256) void k2_if1(
    const unsigned short* __restrict__ y, const float* __restrict__ w2q,
    const float* g2, const float* b2, const float* m2, const float* v2,
    unsigned short* __restrict__ s1) {
  int idx = blockIdx.x * 256 + threadIdx.x;       // (b, f2, wp)
  if (idx >= BATCH * F2C * 200) return;
  int wp = idx % 200, f2 = (idx / 200) % F2C, b = idx / (200 * F2C);
  int g = f2 / 3;
  float inv = g2[f2] * rsqrtf(v2[f2] + 1e-5f);
  float sh  = b2[f2] - m2[f2] * inv;
  const unsigned short* yb = y + (((size_t)b * F1C + g) * CHN) * WOUT;
  float cur[4];
  #pragma unroll
  for (int j = 0; j < 4; ++j) {
    int w = wp * 4 + j;
    float s = 0.f;
    for (int c = 0; c < CHN; ++c) s += bf2f(yb[c * WOUT + w]) * w2q[f2 * CHN + c];
    cur[j] = s * inv + sh;
  }
  float mem[4] = {0.f, 0.f, 0.f, 0.f};
  for (int t = 0; t < TST; ++t) {
    float a = 0.f;
    #pragma unroll
    for (int j = 0; j < 4; ++j) {
      mem[j] += cur[j];
      float sp = (mem[j] >= 1.0f) ? 1.0f : 0.f;
      mem[j] -= sp; a += sp;
    }
    size_t base = (((size_t)t * BATCH + b) * F2C + f2) * S1W;
    s1[base + 12 + wp] = f2bf(a * 0.25f);
    if (wp < 12) { s1[base + wp] = 0; s1[base + 212 + wp] = 0; }
  }
}

// ===== K3: per-batch temporal loop: depthwise + pointwise(WMMA) + IF + FC ==
__global__ __launch_bounds__(256) void k3_stage2(
    const unsigned short* __restrict__ s1,
    const float* __restrict__ w3q, const float* __restrict__ w4q,
    const float* g3, const float* b3, const float* m3, const float* v3,
    const float* __restrict__ wfc, const float* __restrict__ bfc,
    float* __restrict__ out) {
  __shared__ unsigned short s1t[F2C][S1W];  // s1[t,b] tile incl. zero halos
  __shared__ unsigned short za[208][32];    // depthwise out, A-matrix [w][f2in]
  __shared__ float memv[208][32];           // membrane state
  __shared__ float w3s[F2C][25];
  __shared__ unsigned Bpfrag[2][32][8];     // pre-packed pointwise B fragments
  __shared__ float inv3[32], sh3[32];
  __shared__ float accred[4];

  const int b = blockIdx.x, tid = threadIdx.x;
  const int wave = tid >> 5, lane = tid & 31, half = lane >> 4, lrow = lane & 15;

  for (int i = tid; i < 208 * 32; i += 256) ((float*)memv)[i] = 0.f;
  for (int i = tid; i < F2C * 25; i += 256) ((float*)w3s)[i] = w3q[i];
  for (int i = tid; i < 2 * 32 * 8; i += 256) {
    int nt = i >> 8, ln = (i >> 3) & 31, j = i & 7;
    int nn = nt * 16 + (ln & 15), K = (ln >> 4) * 16 + 2 * j;
    unsigned lo = (nn < F2C && K     < F2C) ? (unsigned)f2bf(w4q[nn * F2C + K])     : 0u;
    unsigned hi = (nn < F2C && K + 1 < F2C) ? (unsigned)f2bf(w4q[nn * F2C + K + 1]) : 0u;
    Bpfrag[nt][ln][j] = lo | (hi << 16);
  }
  if (tid < 32) {
    if (tid < F2C) {
      float iv = g3[tid] * rsqrtf(v3[tid] + 1e-5f);
      inv3[tid] = iv; sh3[tid] = b3[tid] - m3[tid] * iv;
    } else { inv3[tid] = 0.f; sh3[tid] = 0.f; }
  }
  if (tid < 4) accred[tid] = 0.f;

  float acc0 = 0.f, acc1 = 0.f, acc2 = 0.f, acc3 = 0.f;
  const unsigned lds_s1t = (unsigned)(size_t)(&s1t[0][0]);   // LDS byte offset

  for (int t = 0; t < TST; ++t) {
    __syncthreads();
    const unsigned short* s1b = s1 + (((size_t)t * BATCH + b) * F2C) * S1W;
#if __has_builtin(__builtin_amdgcn_tensor_load_to_lds) && __has_builtin(__builtin_amdgcn_s_wait_tensorcnt)
    if (wave == 0) {
      // Tensor DMA: 1D tile of 30*224 bf16 (13440 B), global -> LDS.
      unsigned long long ga = (unsigned long long)(size_t)s1b;
      u32x4 g0; i32x8 g1;
      i32x4 gz4 = {0, 0, 0, 0};
      i32x8 gz8 = {0, 0, 0, 0, 0, 0, 0, 0};
      g0.x = 1u;                                    // count=1 valid descriptor
      g0.y = lds_s1t;                               // lds_addr
      g0.z = (unsigned)(ga & 0xFFFFFFFFull);        // global_addr[31:0]
      g0.w = (unsigned)((ga >> 32) & 0x01FFFFFFull) | 0x80000000u; // [56:32] | type=2
      g1.s0 = 0x00010000;                           // mask=0, data_size=2B
      g1.s1 = (int)((30u * S1W & 0xFFFFu) << 16);   // tensor_dim0 lo16
      g1.s2 = (int)((30u * S1W) >> 16 | (1u << 16));// tensor_dim0 hi16, tensor_dim1=1
      g1.s3 = (int)((30u * S1W & 0xFFFFu) << 16);   // tile_dim0 = 6720
      g1.s4 = 0;                                    // tile_dim1/2 unused
      g1.s5 = (int)(30u * S1W);                     // tensor_dim0_stride
      g1.s6 = 0; g1.s7 = 0;
      __builtin_amdgcn_tensor_load_to_lds(g0, g1, gz4, gz4, gz8, 0);
      __builtin_amdgcn_s_wait_tensorcnt(0);
    }
#else
    for (int i = tid; i < F2C * S1W; i += 256) ((unsigned short*)s1t)[i] = s1b[i];
#endif
    if (t + 1 < TST) {   // prefetch next timestep's tile into L2/L0
      const unsigned short* nxt = s1 + (((size_t)(t + 1) * BATCH + b) * F2C) * S1W;
      __builtin_prefetch(nxt + (size_t)tid * 32, 0, 1);
    }
    __syncthreads();
    // depthwise (1,25): za[w][f] = sum_k s1t[f][w+k] * w3s[f][k]
    for (int i = tid; i < 208 * 32; i += 256) {
      int w = i >> 5, f = i & 31;
      float z = 0.f;
      if (w < 200 && f < F2C) {
        #pragma unroll
        for (int k = 0; k < 25; ++k) z += bf2f(s1t[f][w + k]) * w3s[f][k];
      }
      za[w][f] = f2bf(z);
    }
    __syncthreads();
    // pointwise 30x30 as 13 M-tiles x 2 N-tiles of v_wmma_f32_16x16x32_bf16
    for (int job = wave; job < 26; job += 8) {
      int mt = job % 13, nt = job / 13;
      AFrag a, bm;
      const unsigned* arow = (const unsigned*)&za[mt * 16 + lrow][0];
      #pragma unroll
      for (int j = 0; j < 8; ++j) {
        int dw = half * 8 + j;
        a.u32[j] = arow[dw];
      }
      const unsigned* bp = &Bpfrag[nt][lane][0];
      #pragma unroll
      for (int j = 0; j < 8; ++j) bm.u32[j] = bp[j];
      v8f c0 = {0.f,0.f,0.f,0.f,0.f,0.f,0.f,0.f};
      v8f d = __builtin_amdgcn_wmma_f32_16x16x32_bf16(false, a.v, false, bm.v,
                                                      (short)0, c0, false, false);
      int n = nt * 16 + lrow;
      if (n < F2C) {
        float iv = inv3[n], sh = sh3[n];
        #pragma unroll
        for (int r = 0; r < 8; ++r) {
          int w = mt * 16 + r + half * 8;
          float mm = memv[w][n] + d[r] * iv + sh;
          float sp = (mm >= 1.0f) ? 1.0f : 0.f;
          memv[w][n] = mm - sp;
          if (sp > 0.f && w < 198) {              // pool3 keeps w 0..197
            int col = n * 66 + w / 3;
            acc0 += wfc[0 * 1980 + col];
            acc1 += wfc[1 * 1980 + col];
            acc2 += wfc[2 * 1980 + col];
            acc3 += wfc[3 * 1980 + col];
          }
        }
      }
    }
  }
  __syncthreads();
  atomicAdd(&accred[0], acc0);
  atomicAdd(&accred[1], acc1);
  atomicAdd(&accred[2], acc2);
  atomicAdd(&accred[3], acc3);
  __syncthreads();
  if (tid < 4) out[b * 4 + tid] = accred[tid] * (1.0f / 45.0f) + bfc[tid];
}

// ============================ launch ============================
extern "C" void kernel_launch(void* const* d_in, const int* in_sizes, int n_in,
                              void* d_out, int out_size, void* d_ws, size_t ws_size,
                              hipStream_t stream) {
  (void)in_sizes; (void)n_in; (void)out_size; (void)ws_size;
  const float* x    = (const float*)d_in[0];
  const float* w1   = (const float*)d_in[1];
  const float* w2   = (const float*)d_in[2];
  const float* w3   = (const float*)d_in[3];
  const float* w4   = (const float*)d_in[4];
  const float* wfc  = (const float*)d_in[5];
  const float* bfc  = (const float*)d_in[6];
  const float* g1 = (const float*)d_in[7],  *b1 = (const float*)d_in[8];
  const float* m1 = (const float*)d_in[9],  *v1 = (const float*)d_in[10];
  const float* g2 = (const float*)d_in[11], *b2 = (const float*)d_in[12];
  const float* m2 = (const float*)d_in[13], *v2 = (const float*)d_in[14];
  const float* g3 = (const float*)d_in[15], *b3 = (const float*)d_in[16];
  const float* m3 = (const float*)d_in[17], *v3 = (const float*)d_in[18];

  char* ws = (char*)d_ws;
  float* w2q = (float*)(ws + WS_W2Q);
  float* w3q = (float*)(ws + WS_W3Q);
  float* w4q = (float*)(ws + WS_W4Q);
  unsigned short* y  = (unsigned short*)(ws + WS_Y);
  unsigned short* s1 = (unsigned short*)(ws + WS_S1);
  float* outp = (float*)d_out;

  k0_quant<<<1, 256, 0, stream>>>(w2, w3, w4, w2q, w3q, w4q);
  k1_conv1<<<dim3(13, CHN, BATCH), 128, 0, stream>>>(x, w1, g1, b1, m1, v1, y);
  k2_if1<<<(BATCH * F2C * 200) / 256, 256, 0, stream>>>(y, w2q, g2, b2, m2, v2, s1);
  k3_stage2<<<BATCH, 256, 0, stream>>>(s1, w3q, w4q, g3, b3, m3, v3, wfc, bfc, outp);
}
